// BernoulliDecompAttModel_15135464751318
// MI455X (gfx1250) — compile-verified, matrix operand order checked by reading, not run on hardware
//
#include <hip/hip_runtime.h>
#include <stdint.h>

// ---------------------------------------------------------------------------
// Model dims (compile-time constants from the reference)
// ---------------------------------------------------------------------------
#define B_     64
#define L_     256
#define E_     512
#define D_     512
#define TWO_E  1024
#define FOUR_E 2048
#define ML_    (B_ * L_)      // 16384 rows when batch folds into M
#define NEG_BIG (-1e9f)

typedef unsigned short u16;
typedef __attribute__((ext_vector_type(16))) __bf16 bf16x16;
typedef __attribute__((ext_vector_type(8)))  float  fx8;

union FragAB { bf16x16 v; uint4 u[2]; };

__device__ __forceinline__ u16 f2bf(float f) {
    uint32_t x = __float_as_uint(f);
    uint32_t r = x + 0x7FFFu + ((x >> 16) & 1u);   // round-to-nearest-even
    return (u16)(r >> 16);
}

// ---------------------------------------------------------------------------
// Async global->LDS staging (CDNA5 ASYNCcnt path) via inline asm.
// GLOBAL_LOAD_ASYNC_TO_LDS_B128 (VFLAT/VGLOBAL enc): VDST = LDS byte address,
// VADDR = 64-bit global address ("GV" mode). Generic LDS pointer's low 32
// bits are the LDS byte offset (ISA aperture rule), so a truncating cast is
// the correct VDST payload.
// ---------------------------------------------------------------------------
__device__ __forceinline__ void ld128_lds(const u16* gp, u16* lp, bool valid)
{
    if (valid) {
        uint32_t lds_addr = (uint32_t)(uintptr_t)lp;
        asm volatile("global_load_async_to_lds_b128 %0, %1, off"
                     :
                     : "v"(lds_addr), "v"(gp)
                     : "memory");
    } else {
        uint4 z = {0u, 0u, 0u, 0u};
        *(uint4*)lp = z;                 // zero-fill OOB rows (partial M tile)
    }
}

__device__ __forceinline__ void async_fence()
{
    // wait for this wave's outstanding async LDS writes (ASYNCcnt)
    asm volatile("s_wait_asynccnt 0x0" ::: "memory");
}

// ---------------------------------------------------------------------------
// WMMA GEMM:  C[M,N] = epilogue( A[M,K](bf16) x B ), f32 accumulate.
//   B_KN=false : B stored [N,K] (pre-transposed weights / q-as-keys), K contig,
//                staged with async global->LDS copies (no VGPR bounce).
//   B_KN=true  : B stored [K,N] (activations, e.g. ctx = att @ x); transposed
//                into LDS during store (sync path + global_prefetch).
//   Block tile 128(M) x 128(N), 256 threads = 8 wave32, K-step 32,
//   double-buffered LDS; each wave owns 32x64 = 2x4 v_wmma_f32_16x16x32_bf16.
// ---------------------------------------------------------------------------
template<bool B_KN, bool RELU, bool BIAS, bool OUT_BF, bool OUT_F32>
__global__ __launch_bounds__(256)
void gemm_wmma(const u16* __restrict__ A, long lda, long bsA,
               const u16* __restrict__ Bm, long ldb, long bsB,
               const float* __restrict__ bias,
               u16* __restrict__ Cb, float* __restrict__ Cf,
               long ldc, long bsC, int M, int N, int K)
{
    // rows padded to 48 shorts (96B): keeps b128 LDS accesses 16B aligned
    __shared__ u16 As[2][128 * 48];
    __shared__ u16 Bs[2][128 * 48];

    const int tid   = threadIdx.x;
    const int lane  = tid & 31;
    const int wave  = tid >> 5;       // 0..7
    const int wm    = wave & 3;       // 4 waves x 32 rows
    const int wn    = wave >> 2;      // 2 waves x 64 cols
    const int tileN = blockIdx.x * 128;
    const int tileM = blockIdx.y * 128;
    const int batch = blockIdx.z;

    A  += (long)batch * bsA;
    Bm += (long)batch * bsB;

    // per-thread staging coordinates (constant across K-steps)
    int  arow[2], akc[2];  bool aval[2];
    #pragma unroll
    for (int it = 0; it < 2; ++it) {
        int c    = tid + it * 256;          // 512 chunks of 8 shorts: 128x32
        arow[it] = c >> 2;
        akc[it]  = (c & 3) * 8;
        aval[it] = (tileM + arow[it]) < M;
    }
    int  brow[2], bkc[2];  bool bval[2];
    #pragma unroll
    for (int it = 0; it < 2; ++it) {
        int c = tid + it * 256;
        if (B_KN) {                         // 32(K) x 128(N): transpose on store
            brow[it] = c >> 4;              // kk 0..31
            bkc[it]  = (c & 15) * 8;        // n-chunk 0..120
            bval[it] = (tileN + bkc[it]) < N;
        } else {                            // 128(N) x 32(K): same as A
            brow[it] = c >> 2;
            bkc[it]  = (c & 3) * 8;
            bval[it] = (tileN + brow[it]) < N;
        }
    }

    fx8 acc[2][4];
    {
        fx8 z = {0.f,0.f,0.f,0.f,0.f,0.f,0.f,0.f};
        #pragma unroll
        for (int i = 0; i < 2; ++i)
            #pragma unroll
            for (int j = 0; j < 4; ++j) acc[i][j] = z;
    }

    const int lr   = lane & 15;        // fragment row/col within 16
    const int koff = (lane >> 4) * 8;  // K-half select (ISA A/B layout)

    auto stage = [&](int buf, int k0) {
        #pragma unroll
        for (int it = 0; it < 2; ++it)
            ld128_lds(A + (long)(tileM + arow[it]) * lda + (k0 + akc[it]),
                      &As[buf][arow[it] * 48 + akc[it]], aval[it]);
        if (B_KN) {
            #pragma unroll
            for (int it = 0; it < 2; ++it) {
                uint4 v = {0u, 0u, 0u, 0u};
                if (bval[it]) {
                    const u16* bp = Bm + (long)(k0 + brow[it]) * ldb
                                       + (tileN + bkc[it]);
                    v = *(const uint4*)bp;
                    __builtin_prefetch(bp + 32 * ldb, 0, 1);  // next K-slice
                }
                const u16* pv = (const u16*)&v;
                #pragma unroll
                for (int j = 0; j < 8; ++j)
                    Bs[buf][(bkc[it] + j) * 48 + brow[it]] = pv[j];
            }
        } else {
            #pragma unroll
            for (int it = 0; it < 2; ++it)
                ld128_lds(Bm + (long)(tileN + brow[it]) * ldb + (k0 + bkc[it]),
                          &Bs[buf][brow[it] * 48 + bkc[it]], bval[it]);
        }
    };

    // ---- pipelined main loop: stage k+1 async while computing k ----
    stage(0, 0);
    async_fence();
    __syncthreads();

    for (int k0 = 0; k0 < K; k0 += 32) {
        const int cur = (k0 >> 5) & 1;
        if (k0 + 32 < K) stage(cur ^ 1, k0 + 32);

        FragAB fa[2], fb[4];
        #pragma unroll
        for (int i = 0; i < 2; ++i) {
            int r = wm * 32 + i * 16 + lr;
            fa[i].u[0] = *(const uint4*)(&As[cur][r * 48 + koff]);
            fa[i].u[1] = *(const uint4*)(&As[cur][r * 48 + 16 + koff]);
        }
        #pragma unroll
        for (int j = 0; j < 4; ++j) {
            int r = wn * 64 + j * 16 + lr;
            fb[j].u[0] = *(const uint4*)(&Bs[cur][r * 48 + koff]);
            fb[j].u[1] = *(const uint4*)(&Bs[cur][r * 48 + 16 + koff]);
        }
        #pragma unroll
        for (int i = 0; i < 2; ++i)
            #pragma unroll
            for (int j = 0; j < 4; ++j)
                acc[i][j] = __builtin_amdgcn_wmma_f32_16x16x32_bf16(
                    false, fa[i].v, false, fb[j].v,
                    (short)0, acc[i][j], false, false);

        async_fence();       // next-tile LDS writes (ASYNCcnt) done
        __syncthreads();
    }

    // ---- epilogue: bias + ReLU, write bf16 and/or f32 ----
    const int mofs = (lane < 16) ? 0 : 8;   // C layout: VGPR r -> M=r / r+8
    #pragma unroll
    for (int i = 0; i < 2; ++i) {
        #pragma unroll
        for (int j = 0; j < 4; ++j) {
            int n = tileN + wn * 64 + j * 16 + lr;
            float bv = 0.0f;
            if (BIAS && n < N) bv = bias[n];
            #pragma unroll
            for (int r = 0; r < 8; ++r) {
                int m = tileM + wm * 32 + i * 16 + mofs + r;
                if (m < M && n < N) {
                    float v = acc[i][j][r];
                    if (BIAS) v += bv;
                    if (RELU) v = v > 0.0f ? v : 0.0f;
                    long off = (long)batch * bsC + (long)m * ldc + n;
                    if (OUT_BF)  Cb[off] = f2bf(v);
                    if (OUT_F32) Cf[off] = v;
                }
            }
        }
    }
}

// ---------------------------------------------------------------------------
// Embedding gather + pad mask (writes bf16 x into column slice of cat buffer)
// ---------------------------------------------------------------------------
__global__ __launch_bounds__(256)
void embed_kernel(const int* __restrict__ tok, const float* __restrict__ EW,
                  u16* __restrict__ dst, float* __restrict__ mask, long ldc)
{
    int bq = blockIdx.x;                    // b*L + l
    int t  = tok[bq];
    if (threadIdx.x == 0) mask[bq] = (t != 0) ? 1.0f : 0.0f;
    const float* row = EW + (long)t * E_;
    for (int e = threadIdx.x; e < E_; e += 256)
        dst[(long)bq * ldc + e] = f2bf(row[e]);
}

// Transpose fp32 weight [in,out] -> bf16 [out,in] (K contiguous for WMMA B)
__global__ void wconv(const float* __restrict__ W, u16* __restrict__ Wt,
                      int in_dim, int out_dim)
{
    int idx = blockIdx.x * 256 + threadIdx.x;
    if (idx < in_dim * out_dim) {
        int n = idx / in_dim;
        int k = idx - n * in_dim;
        Wt[idx] = f2bf(W[(long)k * out_dim + n]);
    }
}

// ---------------------------------------------------------------------------
// Self-attention softmax: +rel-dist bias, diag mask, key pad mask. One block
// per (b,q) row, 256 threads = 256 keys, LDS tree reduction.
// ---------------------------------------------------------------------------
__global__ __launch_bounds__(256)
void self_softmax(const float* __restrict__ sc, u16* __restrict__ att,
                  const float* __restrict__ mask, const float* __restrict__ distW)
{
    __shared__ float red[256];
    int bq = blockIdx.x;            // b*L + q
    int b  = bq >> 8;
    int q  = bq & 255;
    int k  = threadIdx.x;

    int d = k - q; d = d < -11 ? -11 : (d > 11 ? 11 : d);
    float v = sc[(long)bq * 256 + k] + distW[d + 11];
    if (k == q) v = NEG_BIG;
    if (mask[b * 256 + k] <= 0.0f) v = NEG_BIG;

    red[k] = v; __syncthreads();
    for (int s = 128; s > 0; s >>= 1) {
        if (k < s) red[k] = fmaxf(red[k], red[k + s]);
        __syncthreads();
    }
    float mx = red[0]; __syncthreads();
    float e = __expf(v - mx);
    red[k] = e; __syncthreads();
    for (int s = 128; s > 0; s >>= 1) {
        if (k < s) red[k] += red[k + s];
        __syncthreads();
    }
    att[(long)bq * 256 + k] = f2bf(e / red[0]);
}

// Inter-attention masked softmax; rs/cs allow reading z or z^T.
__global__ __launch_bounds__(256)
void inter_softmax(const float* __restrict__ z, u16* __restrict__ att,
                   const float* __restrict__ kmask, int rs, int cs)
{
    __shared__ float red[256];
    int bq = blockIdx.x;
    int b  = bq >> 8;
    int q  = bq & 255;
    int k  = threadIdx.x;

    float v = (kmask[b * 256 + k] > 0.0f)
                  ? z[(long)b * 65536 + (long)q * rs + (long)k * cs]
                  : NEG_BIG;
    red[k] = v; __syncthreads();
    for (int s = 128; s > 0; s >>= 1) {
        if (k < s) red[k] = fmaxf(red[k], red[k + s]);
        __syncthreads();
    }
    float mx = red[0]; __syncthreads();
    float e = __expf(v - mx);
    red[k] = e; __syncthreads();
    for (int s = 128; s > 0; s >>= 1) {
        if (k < s) red[k] += red[k + s];
        __syncthreads();
    }
    att[(long)bq * 256 + k] = f2bf(e / red[0]);
}

// Masked sum over L: [B,L,D] f32 * mask -> bf16 slice of aggregate input
__global__ __launch_bounds__(256)
void masked_sum(const float* __restrict__ x, const float* __restrict__ mask,
                u16* __restrict__ out, long out_ld, long out_off)
{
    int b = blockIdx.x;
    for (int d = threadIdx.x; d < D_; d += 256) {
        float s = 0.0f;
        for (int l = 0; l < L_; ++l)
            s += x[((long)(b * L_ + l)) * D_ + d] * mask[b * L_ + l];
        out[(long)b * out_ld + out_off + d] = f2bf(s);
    }
}

// Final tiny projection: out[64,3] = agg[64,512] @ Wo[512,3] (fp32)
__global__ void final_out(const float* __restrict__ g, const float* __restrict__ Wo,
                          float* __restrict__ out)
{
    int i = blockIdx.x * 64 + threadIdx.x;
    if (i < B_ * 3) {
        int b = i / 3, o = i - b * 3;
        float s = 0.0f;
        for (int d = 0; d < D_; ++d) s += g[b * D_ + d] * Wo[d * 3 + o];
        out[i] = s;
    }
}

// ---------------------------------------------------------------------------
// Host-side launch helpers (template combos)
// ---------------------------------------------------------------------------
static void g_mlp_bf(hipStream_t s, const u16* A, long lda, const u16* Wt,
                     const float* bias, u16* C, long ldc, int M, int N, int K)
{
    dim3 g((N + 127) / 128, (M + 127) / 128, 1), b(256);
    gemm_wmma<false, true, true, true, false><<<g, b, 0, s>>>(
        A, lda, 0L, Wt, (long)K, 0L, bias, C, (float*)nullptr, ldc, 0L, M, N, K);
}
static void g_mlp_f32(hipStream_t s, const u16* A, long lda, const u16* Wt,
                      const float* bias, float* C, long ldc, int M, int N, int K)
{
    dim3 g((N + 127) / 128, (M + 127) / 128, 1), b(256);
    gemm_wmma<false, true, true, false, true><<<g, b, 0, s>>>(
        A, lda, 0L, Wt, (long)K, 0L, bias, (u16*)nullptr, C, ldc, 0L, M, N, K);
}
static void g_scores(hipStream_t s, const u16* A, long lda, long bsA,
                     const u16* Bnk, long ldb, long bsB,
                     float* C, long ldc, long bsC, int M, int N, int K, int nb)
{
    dim3 g((N + 127) / 128, (M + 127) / 128, nb), b(256);
    gemm_wmma<false, false, false, false, true><<<g, b, 0, s>>>(
        A, lda, bsA, Bnk, ldb, bsB, (const float*)nullptr,
        (u16*)nullptr, C, ldc, bsC, M, N, K);
}
static void g_attmul(hipStream_t s, const u16* A, long lda, long bsA,
                     const u16* Bkn, long ldb, long bsB,
                     u16* C, long ldc, long bsC, int M, int N, int K, int nb)
{
    dim3 g((N + 127) / 128, (M + 127) / 128, nb), b(256);
    gemm_wmma<true, false, false, true, false><<<g, b, 0, s>>>(
        A, lda, bsA, Bkn, ldb, bsB, (const float*)nullptr,
        C, (float*)nullptr, ldc, bsC, M, N, K);
}

// ---------------------------------------------------------------------------
extern "C" void kernel_launch(void* const* d_in, const int* in_sizes, int n_in,
                              void* d_out, int out_size, void* d_ws, size_t ws_size,
                              hipStream_t stream)
{
    (void)in_sizes; (void)n_in; (void)out_size; (void)ws_size;

    const int*   prem  = (const int*)d_in[0];
    const int*   hypo  = (const int*)d_in[1];
    const float* EW    = (const float*)d_in[2];
    const float* distW = (const float*)d_in[3];
    const float* Ws1 = (const float*)d_in[4];  const float* bs1 = (const float*)d_in[5];
    const float* Ws2 = (const float*)d_in[6];  const float* bs2 = (const float*)d_in[7];
    const float* Wa1 = (const float*)d_in[8];  const float* ba1 = (const float*)d_in[9];
    const float* Wa2 = (const float*)d_in[10]; const float* ba2 = (const float*)d_in[11];
    const float* Wc1 = (const float*)d_in[12]; const float* bc1 = (const float*)d_in[13];
    const float* Wc2 = (const float*)d_in[14]; const float* bc2 = (const float*)d_in[15];
    const float* Wg1 = (const float*)d_in[16]; const float* bg1 = (const float*)d_in[17];
    const float* Wg2 = (const float*)d_in[18]; const float* bg2 = (const float*)d_in[19];
    const float* Wo  = (const float*)d_in[20];
    float* out = (float*)d_out;

    // ------- workspace carve-out (256B aligned slices) -------
    char*  ws  = (char*)d_ws;
    size_t off = 0;
    auto carve = [&](size_t bytes) -> char* {
        char* p = ws + off;
        off += (bytes + 255) & ~(size_t)255;
        return p;
    };
    u16* wt_s1 = (u16*)carve((size_t)D_ * E_ * 2);
    u16* wt_s2 = (u16*)carve((size_t)D_ * D_ * 2);
    u16* wt_a1 = (u16*)carve((size_t)D_ * TWO_E * 2);
    u16* wt_a2 = (u16*)carve((size_t)D_ * D_ * 2);
    u16* wt_c1 = (u16*)carve((size_t)D_ * FOUR_E * 2);
    u16* wt_c2 = (u16*)carve((size_t)D_ * D_ * 2);
    u16* wt_g1 = (u16*)carve((size_t)D_ * TWO_E * 2);
    u16* wt_g2 = (u16*)carve((size_t)D_ * D_ * 2);

    u16* cat_p = (u16*)carve((size_t)ML_ * FOUR_E * 2);  // [x|ctx|att_h] cols
    u16* cat_h = (u16*)carve((size_t)ML_ * FOUR_E * 2);  // [x|ctx|att_p] cols
    u16* t0 = (u16*)carve((size_t)ML_ * D_ * 2);
    u16* t1 = (u16*)carve((size_t)ML_ * D_ * 2);
    u16* t2 = (u16*)carve((size_t)ML_ * D_ * 2);
    float* scf   = (float*)carve((size_t)B_ * L_ * L_ * 4);  // scores / z
    u16*   attb  = (u16*)carve((size_t)B_ * L_ * L_ * 2);
    u16*   attb2 = (u16*)carve((size_t)B_ * L_ * L_ * 2);
    float* cmpf  = (float*)carve((size_t)ML_ * D_ * 4);
    float* mask_p = (float*)carve((size_t)ML_ * 4);
    float* mask_h = (float*)carve((size_t)ML_ * 4);
    u16*   agg_in = (u16*)carve((size_t)B_ * TWO_E * 2);
    float* aggf   = (float*)carve((size_t)B_ * D_ * 4);

    // ------- weights -> bf16 [out,in] -------
    auto wc = [&](const float* W, u16* Wt, int in_dim, int out_dim) {
        int n = in_dim * out_dim;
        wconv<<<(n + 255) / 256, 256, 0, stream>>>(W, Wt, in_dim, out_dim);
    };
    wc(Ws1, wt_s1, E_, D_);     wc(Ws2, wt_s2, D_, D_);
    wc(Wa1, wt_a1, TWO_E, D_);  wc(Wa2, wt_a2, D_, D_);
    wc(Wc1, wt_c1, FOUR_E, D_); wc(Wc2, wt_c2, D_, D_);
    wc(Wg1, wt_g1, TWO_E, D_);  wc(Wg2, wt_g2, D_, D_);

    // ------- embedding + masks (x into cat cols [0,E)) -------
    embed_kernel<<<ML_, 256, 0, stream>>>(prem, EW, cat_p, mask_p, FOUR_E);
    embed_kernel<<<ML_, 256, 0, stream>>>(hypo, EW, cat_h, mask_h, FOUR_E);

    const long AL = 65536;           // L*L per-batch stride
    const long XS = (long)L_ * FOUR_E;

    // ------- self branch (prem then hypo) -------
    for (int s = 0; s < 2; ++s) {
        u16*   cat = s ? cat_h : cat_p;
        float* msk = s ? mask_h : mask_p;
        g_mlp_bf(stream, cat, FOUR_E, wt_s1, bs1, t0, D_, ML_, D_, E_);
        g_mlp_bf(stream, t0, D_, wt_s2, bs2, t1, D_, ML_, D_, D_);
        // scores = q q^T  (q rows are keys -> B is [N,K])
        g_scores(stream, t1, D_, (long)L_ * D_, t1, D_, (long)L_ * D_,
                 scf, L_, AL, L_, L_, D_, B_);
        self_softmax<<<ML_, 256, 0, stream>>>(scf, attb, msk, distW);
        // ctx = att @ x  (x is [K=L, N=E] slice of cat, ldb = 4E)
        g_attmul(stream, attb, L_, AL, cat, FOUR_E, XS,
                 cat + E_, FOUR_E, XS, L_, E_, L_, B_);
    }

    // ------- inter attention -------
    g_mlp_bf(stream, cat_p, FOUR_E, wt_a1, ba1, t0, D_, ML_, D_, TWO_E);
    g_mlp_bf(stream, t0, D_, wt_a2, ba2, t1, D_, ML_, D_, D_);   // pq
    g_mlp_bf(stream, cat_h, FOUR_E, wt_a1, ba1, t0, D_, ML_, D_, TWO_E);
    g_mlp_bf(stream, t0, D_, wt_a2, ba2, t2, D_, ML_, D_, D_);   // hk
    g_scores(stream, t1, D_, (long)L_ * D_, t2, D_, (long)L_ * D_,
             scf, L_, AL, L_, L_, D_, B_);                        // z
    inter_softmax<<<ML_, 256, 0, stream>>>(scf, attb,  mask_h, 256, 1);  // p2h
    inter_softmax<<<ML_, 256, 0, stream>>>(scf, attb2, mask_p, 1, 256);  // h2p
    // attended_hypo = p2h @ cat_h[:, :2E] -> cat_p cols [2E,4E)
    g_attmul(stream, attb, L_, AL, cat_h, FOUR_E, XS,
             cat_p + TWO_E, FOUR_E, XS, L_, TWO_E, L_, B_);
    // attended_prem = h2p @ cat_p[:, :2E] -> cat_h cols [2E,4E)
    g_attmul(stream, attb2, L_, AL, cat_p, FOUR_E, XS,
             cat_h + TWO_E, FOUR_E, XS, L_, TWO_E, L_, B_);

    // ------- compare MLP + masked sum -------
    g_mlp_bf(stream, cat_p, FOUR_E, wt_c1, bc1, t0, D_, ML_, D_, FOUR_E);
    g_mlp_f32(stream, t0, D_, wt_c2, bc2, cmpf, D_, ML_, D_, D_);
    masked_sum<<<B_, 256, 0, stream>>>(cmpf, mask_p, agg_in, TWO_E, 0);
    g_mlp_bf(stream, cat_h, FOUR_E, wt_c1, bc1, t0, D_, ML_, D_, FOUR_E);
    g_mlp_f32(stream, t0, D_, wt_c2, bc2, cmpf, D_, ML_, D_, D_);
    masked_sum<<<B_, 256, 0, stream>>>(cmpf, mask_h, agg_in, TWO_E, D_);

    // ------- aggregate + output -------
    g_mlp_bf(stream, agg_in, TWO_E, wt_g1, bg1, t0, D_, B_, D_, TWO_E);
    g_mlp_f32(stream, t0, D_, wt_g2, bg2, aggf, D_, B_, D_, D_);
    final_out<<<(B_ * 3 + 63) / 64, 64, 0, stream>>>(aggf, Wo, out);
}
